// FeatureEncoder_56822417326742
// MI455X (gfx1250) — matrix operand order repaired and checked by reading, not compiled, not run on hardware
//
#include <hip/hip_runtime.h>
#include <cstdint>
#include <cstddef>

#define Hdim 128
#define Fdim 128
#define Ldim 16
#define NSTEP 8
#define DT 0.125f
#define WPITCH 136   // bf16 elems per LDS stage row (128+8 pad) -> 272B rows, 16B aligned

typedef __attribute__((ext_vector_type(16))) __bf16 v16bf;
typedef __attribute__((ext_vector_type(2)))  __bf16 v2bf;
typedef __attribute__((ext_vector_type(8)))  float  v8f;

union FragU { uint4 q[2]; v16bf v; };

// native f32 -> bf16 (RTNE); gfx1250 has hardware BF16 converts
__device__ __forceinline__ uint16_t f2bf(float f) {
    __bf16 b = (__bf16)f;
    return __builtin_bit_cast(uint16_t, b);
}
__device__ __forceinline__ uint32_t pack2(float a, float b) {
    v2bf v; v[0] = (__bf16)a; v[1] = (__bf16)b;
    return __builtin_bit_cast(uint32_t, v);
}
__device__ __forceinline__ float sigm(float x)  { return 1.f / (1.f + __expf(-x)); }
__device__ __forceinline__ float tanh_f(float x){ float e = __expf(2.f * x); return 1.f - 2.f / (e + 1.f); }

// 16-bit A 16x32 fragment (§7.12.2): lane<16 row m K={t*32+0..7, +16..23}; lane>=16 K={+8..15, +24..31}
__device__ __forceinline__ v16bf loadA(const uint16_t* stage, int m, int half, int t) {
    FragU u;
    const char* p = (const char*)stage + (m * WPITCH + t * 32) * 2 + half * 16;
    u.q[0] = *(const uint4*)p;
    u.q[1] = *(const uint4*)(p + 32);
    return u.v;
}
// 16-bit B 32x16 fragment from row-major f32 weight in global memory (B[k][o] = W[o][k])
__device__ __forceinline__ v16bf loadB_gbl(const float* W, int o, int half, int t) {
    const float4* p = (const float4*)(W + o * Fdim + t * 32 + half * 8);
    float4 f0 = p[0], f1 = p[1], f2 = p[4], f3 = p[5];
    FragU u;
    u.q[0].x = pack2(f0.x, f0.y); u.q[0].y = pack2(f0.z, f0.w);
    u.q[0].z = pack2(f1.x, f1.y); u.q[0].w = pack2(f1.z, f1.w);
    u.q[1].x = pack2(f2.x, f2.y); u.q[1].y = pack2(f2.z, f2.w);
    u.q[1].z = pack2(f3.x, f3.y); u.q[1].w = pack2(f3.z, f3.w);
    return u.v;
}

__device__ __forceinline__ v8f wmma_bf16(v16bf a, v16bf b, v8f c) {
    return __builtin_amdgcn_wmma_f32_16x16x32_bf16(false, a, false, b, (short)0, c, false, false);
}

// acc{1,2}[jj] += stage(16x128) @ W{1,2}^T  restricted to this wave's 2 N-tiles, B from global f32
__device__ __forceinline__ void mmg2(const uint16_t* stage, const float* W1, const float* W2,
                                     v8f a1[2], v8f a2[2], int j0, int m, int half) {
    v16bf A[4];
#pragma unroll
    for (int t = 0; t < 4; ++t) A[t] = loadA(stage, m, half, t);
#pragma unroll
    for (int jj = 0; jj < 2; ++jj) {
        int o = (j0 + jj) * 16 + m;
#pragma unroll
        for (int t = 0; t < 4; ++t) {
            a1[jj] = wmma_bf16(A[t], loadB_gbl(W1, o, half, t), a1[jj]);
            a2[jj] = wmma_bf16(A[t], loadB_gbl(W2, o, half, t), a2[jj]);
        }
    }
}
__device__ __forceinline__ void mmg1(const uint16_t* stage, const float* W1,
                                     v8f a1[2], int j0, int m, int half) {
    v16bf A[4];
#pragma unroll
    for (int t = 0; t < 4; ++t) A[t] = loadA(stage, m, half, t);
#pragma unroll
    for (int jj = 0; jj < 2; ++jj) {
        int o = (j0 + jj) * 16 + m;
#pragma unroll
        for (int t = 0; t < 4; ++t)
            a1[jj] = wmma_bf16(A[t], loadB_gbl(W1, o, half, t), a1[jj]);
    }
}

// f(y) = ratio * (1-z) * (tanh((sigmoid(y@Or^T)*y)@On^T) - y);  B operands register-resident
__device__ __forceinline__ void f_eval(const v8f y[2], v8f out[2],
                                       uint16_t* sA, uint16_t* sB,
                                       const v16bf BOr[2][4], const v16bf BOz[2][4], const v16bf BOn[2][4],
                                       const float rat[8], int j0, int m, int half) {
#pragma unroll
    for (int jj = 0; jj < 2; ++jj)
#pragma unroll
        for (int v = 0; v < 8; ++v)
            sA[(v + 8 * half) * WPITCH + (j0 + jj) * 16 + m] = f2bf(y[jj][v]);
    __syncthreads();
    v16bf A[4];
#pragma unroll
    for (int t = 0; t < 4; ++t) A[t] = loadA(sA, m, half, t);
    v8f r[2] = {}, z[2] = {};
#pragma unroll
    for (int jj = 0; jj < 2; ++jj)
#pragma unroll
        for (int t = 0; t < 4; ++t) {
            r[jj] = wmma_bf16(A[t], BOr[jj][t], r[jj]);
            z[jj] = wmma_bf16(A[t], BOz[jj][t], z[jj]);
        }
#pragma unroll
    for (int jj = 0; jj < 2; ++jj)
#pragma unroll
        for (int v = 0; v < 8; ++v) {
            r[jj][v] = sigm(r[jj][v]);
            z[jj][v] = sigm(z[jj][v]);
            sB[(v + 8 * half) * WPITCH + (j0 + jj) * 16 + m] = f2bf(r[jj][v] * y[jj][v]);
        }
    __syncthreads();
    v16bf A2[4];
#pragma unroll
    for (int t = 0; t < 4; ++t) A2[t] = loadA(sB, m, half, t);
    v8f g[2] = {};
#pragma unroll
    for (int jj = 0; jj < 2; ++jj)
#pragma unroll
        for (int t = 0; t < 4; ++t)
            g[jj] = wmma_bf16(A2[t], BOn[jj][t], g[jj]);
#pragma unroll
    for (int jj = 0; jj < 2; ++jj)
#pragma unroll
        for (int v = 0; v < 8; ++v)
            out[jj][v] = (1.f - z[jj][v]) * (tanh_f(g[jj][v]) - y[jj][v]) * rat[v];
}

__global__ void __launch_bounds__(128) __attribute__((amdgpu_waves_per_eu(2)))
gruode_walk_kernel(
    const float* __restrict__ X, const float* __restrict__ Trec,
    const float* __restrict__ Wxr, const float* __restrict__ bxr,
    const float* __restrict__ Wxz, const float* __restrict__ bxz,
    const float* __restrict__ Wxn, const float* __restrict__ bxn,
    const float* __restrict__ Whr, const float* __restrict__ Whz, const float* __restrict__ Whn,
    const float* __restrict__ Or_, const float* __restrict__ Oz_, const float* __restrict__ On_,
    float* __restrict__ out, int N) {
    __shared__ uint16_t sX[16 * WPITCH];   // x_i stage (bf16)
    __shared__ uint16_t sA[16 * WPITCH];   // h / y stage
    __shared__ uint16_t sB[16 * WPITCH];   // r*h stage
    __shared__ float    biasL[3 * 128];
    __shared__ float    ratioL[16];

    const int tid  = threadIdx.x;
    const int wave = tid >> 5, lane = tid & 31;
    const int m = lane & 15, half = lane >> 4;
    const int j0 = wave * 2;                    // this wave's 2 N-tiles (cols 32w..32w+31)
    const int rowbase = blockIdx.x * 16;        // one 16-row tile per block
    if (rowbase >= N) return;

    biasL[tid]       = bxr[tid];
    biasL[128 + tid] = bxz[tid];
    biasL[256 + tid] = bxn[tid];

    // Register-resident B fragments of the hot ODE weights: 3 * 2 * 4 * 8 = 192 VGPRs/lane
    v16bf BOr[2][4], BOz[2][4], BOn[2][4];
#pragma unroll
    for (int jj = 0; jj < 2; ++jj) {
        int o = (j0 + jj) * 16 + m;
#pragma unroll
        for (int t = 0; t < 4; ++t) {
            BOr[jj][t] = loadB_gbl(Or_, o, half, t);
            BOz[jj][t] = loadB_gbl(Oz_, o, half, t);
            BOn[jj][t] = loadB_gbl(On_, o, half, t);
        }
    }

    v8f   h[2] = {};    // this wave's slice of the 16x128 state, WMMA C-fragment layout
    float rat[8];

    // Local (launder-able) copies of the cold GRU weight pointers
    const float* pWxr = Wxr; const float* pWxz = Wxz; const float* pWxn = Wxn;
    const float* pWhr = Whr; const float* pWhz = Whz; const float* pWhn = Whn;

    for (int i = 0; i < Ldim; ++i) {
        // Launder the cold weight pointers so their fragment loads are NOT hoisted out of
        // the loop into ~384 live VGPRs; they re-stream from L1/L2 once per transition.
        asm volatile("" : "+s"(pWxr), "+s"(pWxz), "+s"(pWxn), "+s"(pWhr), "+s"(pWhz), "+s"(pWhn));

        // cooperative x_i stage: thread -> row tid&15, cols (tid>>4)*16 .. +15
        {
            const int xrow = tid & 15, cb = (tid >> 4) * 16;
            const float* xr = X + ((size_t)(rowbase + xrow) * Ldim + i) * Fdim + cb;
#pragma unroll
            for (int c = 0; c < 16; c += 4) {
                float4 fv = *(const float4*)(xr + c);
                uint2 p; p.x = pack2(fv.x, fv.y); p.y = pack2(fv.z, fv.w);
                *(uint2*)((char*)sX + xrow * (WPITCH * 2) + (cb + c) * 2) = p;
            }
            if (i + 1 < Ldim) __builtin_prefetch(xr + Fdim, 0, 3);   // global_prefetch next x
        }
        // stage this wave's h columns
#pragma unroll
        for (int jj = 0; jj < 2; ++jj)
#pragma unroll
            for (int v = 0; v < 8; ++v)
                sA[(v + 8 * half) * WPITCH + (j0 + jj) * 16 + m] = f2bf(h[jj][v]);
        // ratio = log10(|t_i - t_{i+1}| + 1) + 0.01 per row
        if (tid < 16 && i + 1 < Ldim) {
            float t0 = Trec[(size_t)(rowbase + tid) * Ldim + i];
            float t1 = Trec[(size_t)(rowbase + tid) * Ldim + i + 1];
            ratioL[tid] = __log2f(fabsf(t0 - t1) + 1.f) * 0.3010299956639812f + 0.01f;
        }
        __syncthreads();
        if (i + 1 < Ldim) {
#pragma unroll
            for (int v = 0; v < 8; ++v) rat[v] = ratioL[v + 8 * half];
        }

        // ---- GRU cell (this wave's 32 output columns) ----
        v8f ar[2], az[2], ag[2];
#pragma unroll
        for (int jj = 0; jj < 2; ++jj) {
            float br = biasL[(j0 + jj) * 16 + m];
            float bz = biasL[128 + (j0 + jj) * 16 + m];
            float bn = biasL[256 + (j0 + jj) * 16 + m];
#pragma unroll
            for (int v = 0; v < 8; ++v) { ar[jj][v] = br; az[jj][v] = bz; ag[jj][v] = bn; }
        }
        mmg2(sX, pWxr, pWxz, ar, az, j0, m, half);
        mmg1(sX, pWxn, ag, j0, m, half);
        mmg2(sA, pWhr, pWhz, ar, az, j0, m, half);
#pragma unroll
        for (int jj = 0; jj < 2; ++jj)
#pragma unroll
            for (int v = 0; v < 8; ++v) {
                ar[jj][v] = sigm(ar[jj][v]);   // r
                az[jj][v] = sigm(az[jj][v]);   // z
                sB[(v + 8 * half) * WPITCH + (j0 + jj) * 16 + m] = f2bf(ar[jj][v] * h[jj][v]);
            }
        __syncthreads();
        mmg1(sB, pWhn, ag, j0, m, half);
#pragma unroll
        for (int jj = 0; jj < 2; ++jj)
#pragma unroll
            for (int v = 0; v < 8; ++v) {
                float g = tanh_f(ag[jj][v]);
                h[jj][v] = az[jj][v] * h[jj][v] + (1.f - az[jj][v]) * g;
            }

        if (i == Ldim - 1) break;

        // ---- ODE: 8 steps of RK4 (3/8 rule), weights in VGPRs ----
        for (int s = 0; s < NSTEP; ++s) {
            v8f k1[2], k2[2], k3[2], y[2];
            f_eval(h, k1, sA, sB, BOr, BOz, BOn, rat, j0, m, half);
#pragma unroll
            for (int jj = 0; jj < 2; ++jj)
#pragma unroll
                for (int v = 0; v < 8; ++v) y[jj][v] = h[jj][v] + (DT / 3.f) * k1[jj][v];
            f_eval(y, k2, sA, sB, BOr, BOz, BOn, rat, j0, m, half);
#pragma unroll
            for (int jj = 0; jj < 2; ++jj)
#pragma unroll
                for (int v = 0; v < 8; ++v) y[jj][v] = h[jj][v] + DT * (k2[jj][v] - k1[jj][v] * (1.f / 3.f));
            f_eval(y, k3, sA, sB, BOr, BOz, BOn, rat, j0, m, half);
#pragma unroll
            for (int jj = 0; jj < 2; ++jj)
#pragma unroll
                for (int v = 0; v < 8; ++v) {
                    y[jj][v]  = h[jj][v] + DT * (k1[jj][v] - k2[jj][v] + k3[jj][v]);
                    k2[jj][v] = k1[jj][v] + 3.f * (k2[jj][v] + k3[jj][v]);     // partial weighted sum
                }
            f_eval(y, k1, sA, sB, BOr, BOz, BOn, rat, j0, m, half);            // k4 -> k1
#pragma unroll
            for (int jj = 0; jj < 2; ++jj)
#pragma unroll
                for (int v = 0; v < 8; ++v)
                    h[jj][v] += DT * 0.125f * (k2[jj][v] + k1[jj][v]);
        }
    }

    // write this wave's columns of the final h ([N, H] row major)
#pragma unroll
    for (int jj = 0; jj < 2; ++jj)
#pragma unroll
        for (int v = 0; v < 8; ++v)
            out[(size_t)(rowbase + v + 8 * half) * Hdim + (j0 + jj) * 16 + m] = h[jj][v];
}

extern "C" void kernel_launch(void* const* d_in, const int* in_sizes, int n_in,
                              void* d_out, int out_size, void* d_ws, size_t ws_size,
                              hipStream_t stream) {
    const float* X   = (const float*)d_in[0];
    const float* T   = (const float*)d_in[1];
    const float* Wxr = (const float*)d_in[2];
    const float* bxr = (const float*)d_in[3];
    const float* Wxz = (const float*)d_in[4];
    const float* bxz = (const float*)d_in[5];
    const float* Wxn = (const float*)d_in[6];
    const float* bxn = (const float*)d_in[7];
    const float* Whr = (const float*)d_in[8];
    const float* Whz = (const float*)d_in[9];
    const float* Whn = (const float*)d_in[10];
    const float* Or_ = (const float*)d_in[11];
    const float* Oz_ = (const float*)d_in[12];
    const float* On_ = (const float*)d_in[13];
    float* out = (float*)d_out;

    const int N = in_sizes[0] / (Ldim * Fdim);   // 12800 rows
    const int blocks = (N + 15) / 16;            // one 16-row tile per 4-wave block

    hipLaunchKernelGGL(gruode_walk_kernel, dim3(blocks), dim3(128), 0, stream,
                       X, T, Wxr, bxr, Wxz, bxz, Wxn, bxn, Whr, Whz, Whn,
                       Or_, Oz_, On_, out, N);
}
